// SpeciesContrastiveLoss_79250736545852
// MI455X (gfx1250) — matrix operand order repaired and checked by reading, not compiled, not run on hardware
//
#include <hip/hip_runtime.h>
#include <hip/hip_bf16.h>
#include <stdint.h>

#define B_ROWS    8192
#define DIM       256
#define NSPECIES  64
#define INV_TAU   14.2857142857142857f   // 1/0.07
#define MARGIN    0.2f

typedef __attribute__((ext_vector_type(16))) __bf16 v16bf;
typedef __attribute__((ext_vector_type(8)))  __bf16 v8bf;
typedef __attribute__((ext_vector_type(8)))  float  v8f;

__device__ __forceinline__ unsigned short f32_to_bf16_rne(float f) {
    unsigned int u = __float_as_uint(f);
    u += 0x7FFFu + ((u >> 16) & 1u);
    return (unsigned short)(u >> 16);
}
__device__ __forceinline__ float bf16_bits_to_f32(unsigned short h) {
    return __uint_as_float(((unsigned int)h) << 16);
}

// ---------------- Kernel 1: f32 -> bf16 hi/lo split for A (species) and B (text)
__global__ __launch_bounds__(256)
void split_bf16_kernel(const float* __restrict__ A, const float* __restrict__ Bm,
                       unsigned short* __restrict__ aHi, unsigned short* __restrict__ aLo,
                       unsigned short* __restrict__ bHi, unsigned short* __restrict__ bLo,
                       int n) {
    int idx = blockIdx.x * blockDim.x + threadIdx.x;
    if (idx >= n) return;
    float a = A[idx];
    unsigned short ah = f32_to_bf16_rne(a);
    unsigned short al = f32_to_bf16_rne(a - bf16_bits_to_f32(ah));
    aHi[idx] = ah; aLo[idx] = al;
    float b = Bm[idx];
    unsigned short bh = f32_to_bf16_rne(b);
    unsigned short bl = f32_to_bf16_rne(b - bf16_bits_to_f32(bh));
    bHi[idx] = bh; bLo[idx] = bl;
}

// ---------------- Kernel 2: zero the global accumulators
__global__ __launch_bounds__(256)
void zero_kernel(float* __restrict__ p, int n) {
    int i = blockIdx.x * blockDim.x + threadIdx.x;
    if (i < n) p[i] = 0.0f;
}

// ---------------- Kernel 3: fused WMMA GEMM tiles + masked exp-sum / row-sum epilogue
// Block = 256 threads = 8 waves, arranged 4 (row tiles) x 2 (col tiles) -> 64x32 block tile.
// Each wave: 16x16 output tile, K=256 in 8 steps of 32, bf16x3 split precision.
__global__ __launch_bounds__(256)
void sim_tile_kernel(const unsigned short* __restrict__ aHiU, const unsigned short* __restrict__ aLoU,
                     const unsigned short* __restrict__ bHiU, const unsigned short* __restrict__ bLoU,
                     const int* __restrict__ ids,
                     float* __restrict__ rowExp, float* __restrict__ colExp,
                     float* __restrict__ rowSum, float* __restrict__ rowSameSum,
                     float* __restrict__ diag) {
    const __bf16* aHi = (const __bf16*)aHiU;
    const __bf16* aLo = (const __bf16*)aLoU;
    const __bf16* bHi = (const __bf16*)bHiU;
    const __bf16* bLo = (const __bf16*)bLoU;

    const int lane = threadIdx.x & 31;
    const int wave = threadIdx.x >> 5;      // 0..7
    const int half = lane >> 4;             // 0 or 1
    const int l16  = lane & 15;

    const int rowBase = blockIdx.y * 64 + (wave >> 1) * 16;
    const int colBase = blockIdx.x * 32 + (wave & 1) * 16;

    // A fragment: lane l16 (half h) holds row rowBase+l16, K groups {koff..koff+7, koff+16..koff+23}
    const int aRow = rowBase + l16;
    const int bCol = colBase + l16;
    const int koffA = half << 3;            // 0 or 8
    const int koffB = half << 3;            // 0 or 8 (K groups {koff..koff+7, koff+16..koff+23}, same as A)

    v8f accHH = {};
    v8f accHL = {};
    v8f accLH = {};

    for (int kk = 0; kk < DIM; kk += 32) {
        const __bf16* pah = aHi + (size_t)aRow * DIM + kk + koffA;
        const __bf16* pal = aLo + (size_t)aRow * DIM + kk + koffA;
        v8bf ah0 = *(const v8bf*)(pah);
        v8bf ah1 = *(const v8bf*)(pah + 16);
        v8bf al0 = *(const v8bf*)(pal);
        v8bf al1 = *(const v8bf*)(pal + 16);
        v16bf afh = __builtin_shufflevector(ah0, ah1, 0,1,2,3,4,5,6,7,8,9,10,11,12,13,14,15);
        v16bf afl = __builtin_shufflevector(al0, al1, 0,1,2,3,4,5,6,7,8,9,10,11,12,13,14,15);

        const __bf16* pbh = bHi + (size_t)bCol * DIM + kk + koffB;
        const __bf16* pbl = bLo + (size_t)bCol * DIM + kk + koffB;
        v8bf bh0 = *(const v8bf*)(pbh);
        v8bf bh1 = *(const v8bf*)(pbh + 16);
        v8bf bl0 = *(const v8bf*)(pbl);
        v8bf bl1 = *(const v8bf*)(pbl + 16);
        v16bf bfh = __builtin_shufflevector(bh0, bh1, 0,1,2,3,4,5,6,7,8,9,10,11,12,13,14,15);
        v16bf bfl = __builtin_shufflevector(bl0, bl1, 0,1,2,3,4,5,6,7,8,9,10,11,12,13,14,15);

        // three independent accumulator chains (avoids WMMA->WMMA RAW stalls)
        accHH = __builtin_amdgcn_wmma_f32_16x16x32_bf16(false, afh, false, bfh,
                                                        (short)0, accHH, false, false);
        accHL = __builtin_amdgcn_wmma_f32_16x16x32_bf16(false, afh, false, bfl,
                                                        (short)0, accHL, false, false);
        accLH = __builtin_amdgcn_wmma_f32_16x16x32_bf16(false, afl, false, bfh,
                                                        (short)0, accLH, false, false);
    }

    // ---------------- epilogue: masked exp-sums + row sums + diagonal
    const int sidj = ids[bCol];
    float colAcc = 0.0f;

    #pragma unroll
    for (int r = 0; r < 8; ++r) {
        const int row = rowBase + (half << 3) + r;  // C/D layout: M = r + 8*half, N = l16
        float s = (accHL[r] + accLH[r]) + accHH[r];
        if (row == bCol) diag[row] = s;             // each (i,i) computed in exactly one tile
        const bool same = (ids[row] == sidj);
        float e  = same ? __expf((s - 1.0f) * INV_TAU) : 0.0f;  // fixed-shift lse term
        colAcc += e;

        float re = e, rs = s, rss = same ? s : 0.0f;
        #pragma unroll
        for (int off = 1; off < 16; off <<= 1) {    // reduce across the 16 lanes sharing this row
            re  += __shfl_xor(re,  off, 32);
            rs  += __shfl_xor(rs,  off, 32);
            rss += __shfl_xor(rss, off, 32);
        }
        if (l16 == 0) {
            atomicAdd(&rowExp[row],     re);
            atomicAdd(&rowSum[row],     rs);
            atomicAdd(&rowSameSum[row], rss);
        }
    }
    // both lane-halves cover the same 16 columns -> combine then one atomic per column
    colAcc += __shfl_xor(colAcc, 16, 32);
    if (half == 0) atomicAdd(&colExp[bCol], colAcc);
}

// ---------------- Kernel 4: species segment sums + CE + triplet + final scalar
__global__ __launch_bounds__(1024)
void finalize_kernel(const float* __restrict__ rowExp, const float* __restrict__ colExp,
                     const float* __restrict__ rowSum, const float* __restrict__ rowSameSum,
                     const float* __restrict__ diag, const int* __restrict__ ids,
                     float* __restrict__ out) {
    __shared__ float cnt[NSPECIES], segv[NSPECIES], segt[NSPECIES];
    __shared__ float triSum, triCnt;
    const int tid = threadIdx.x;
    if (tid < NSPECIES) { cnt[tid] = 0.0f; segv[tid] = 0.0f; segt[tid] = 0.0f; }
    if (tid == 0) { triSum = 0.0f; triCnt = 0.0f; }
    __syncthreads();

    for (int i = tid; i < B_ROWS; i += blockDim.x)
        atomicAdd(&cnt[ids[i]], 1.0f);
    __syncthreads();

    for (int i = tid; i < B_ROWS; i += blockDim.x) {
        const int s = ids[i];
        const float d  = diag[i];
        const float dl = d * INV_TAU;
        // lse = log(sum exp(x - 1/tau)) + 1/tau  (fixed shift; sim <= 1)
        const float cev = __logf(rowExp[i]) + INV_TAU - dl;
        const float cet = __logf(colExp[i]) + INV_TAU - dl;
        atomicAdd(&segv[s], cev);
        atomicAdd(&segt[s], cet);

        const float c        = cnt[s];
        const float pos_cnt  = c - 1.0f;
        const float neg_cnt  = (float)B_ROWS - c;
        const float pos_mean = (rowSameSum[i] - d)         / fmaxf(pos_cnt, 1.0f);
        const float neg_mean = (rowSum[i] - rowSameSum[i]) / fmaxf(neg_cnt, 1.0f);
        const float tri      = fmaxf(neg_mean - pos_mean + MARGIN, 0.0f);
        const bool  valid    = (pos_cnt > 0.0f) && (neg_cnt > 0.0f);
        atomicAdd(&triSum, valid ? tri  : 0.0f);
        atomicAdd(&triCnt, valid ? 1.0f : 0.0f);
    }
    __syncthreads();

    if (tid == 0) {
        float acc = 0.0f, nv = 0.0f;
        for (int s = 0; s < NSPECIES; ++s) {
            if (cnt[s] >= 2.0f) {
                acc += (segv[s] + segt[s]) / (2.0f * fmaxf(cnt[s], 1.0f));
                nv  += 1.0f;
            }
        }
        const float loss_infonce = acc / fmaxf(nv, 1.0f);
        const float loss_triplet = triSum / fmaxf(triCnt, 1.0f);
        out[0] = loss_infonce + loss_triplet;
    }
}

// ---------------- host launcher
extern "C" void kernel_launch(void* const* d_in, const int* in_sizes, int n_in,
                              void* d_out, int out_size, void* d_ws, size_t ws_size,
                              hipStream_t stream) {
    const float* species = (const float*)d_in[0];
    const float* text    = (const float*)d_in[1];
    const int*   ids     = (const int*)d_in[2];
    float* out = (float*)d_out;

    const int n = B_ROWS * DIM;  // 2,097,152 elements per matrix
    // workspace layout: 4x bf16 matrices (16 MB) + 5x f32 vectors (160 KB)
    unsigned short* aHi = (unsigned short*)d_ws;
    unsigned short* aLo = aHi + n;
    unsigned short* bHi = aLo + n;
    unsigned short* bLo = bHi + n;
    float* rowExp     = (float*)(bLo + n);
    float* colExp     = rowExp + B_ROWS;
    float* rowSum     = colExp + B_ROWS;
    float* rowSameSum = rowSum + B_ROWS;
    float* diag       = rowSameSum + B_ROWS;

    split_bf16_kernel<<<(n + 255) / 256, 256, 0, stream>>>(species, text, aHi, aLo, bHi, bLo, n);
    zero_kernel<<<(4 * B_ROWS + 255) / 256, 256, 0, stream>>>(rowExp, 4 * B_ROWS);

    dim3 grid(B_ROWS / 32, B_ROWS / 64);  // 256 x 128 blocks of 64x32 tiles
    sim_tile_kernel<<<grid, 256, 0, stream>>>(aHi, aLo, bHi, bLo, ids,
                                              rowExp, colExp, rowSum, rowSameSum, diag);

    finalize_kernel<<<1, 1024, 0, stream>>>(rowExp, colExp, rowSum, rowSameSum, diag, ids, out);
}
